// AlignmentEncoder_66357244723442
// MI455X (gfx1250) — matrix operand order repaired and checked by reading, not compiled, not run on hardware
//
#include <hip/hip_runtime.h>
#include <hip/hip_bf16.h>
#include <stdint.h>

typedef __bf16 bf16_t;
typedef __attribute__((ext_vector_type(16))) __bf16 v16bf;
typedef __attribute__((ext_vector_type(8)))  __bf16 v8bf;
typedef __attribute__((ext_vector_type(8)))  float  v8f;

#define TEMP_C   0.0005f
#define MASK_VAL -1000000000.0f

// ---------------------------------------------------------------------------
// Prep kernels
// ---------------------------------------------------------------------------
__global__ void k_cast_pad_rows(const float* __restrict__ src, bf16_t* __restrict__ dst,
                                int rows, int wsrc, int wdst) {
  long long i = (long long)blockIdx.x * blockDim.x + threadIdx.x;
  long long total = (long long)rows * wdst;
  if (i >= total) return;
  int c = (int)(i % wdst);
  long long r = i / wdst;
  dst[i] = (c < wsrc) ? (bf16_t)src[r * wsrc + c] : (bf16_t)0.f;
}

__global__ void k_embed(const int* __restrict__ keys, const float* __restrict__ emb,
                        bf16_t* __restrict__ dst, int rows, int C) {
  long long i = (long long)blockIdx.x * blockDim.x + threadIdx.x;
  long long total = (long long)rows * C;
  if (i >= total) return;
  int c = (int)(i % C);
  long long r = i / C;
  dst[i] = (bf16_t)emb[(long long)keys[r] * C + c];
}

// src: [KW, Kdim, Ndim] float  ->  dst: [KW, Nld, Kpad] bf16 TRANSPOSED, zero padded
__global__ void k_cast_weight_t(const float* __restrict__ src, bf16_t* __restrict__ dst,
                                int KW, int Kdim, int Kpad, int Ndim, int Nld) {
  long long i = (long long)blockIdx.x * blockDim.x + threadIdx.x;
  long long total = (long long)KW * Nld * Kpad;
  if (i >= total) return;
  int k    = (int)(i % Kpad);
  long long rest = i / Kpad;
  int n    = (int)(rest % Nld);
  int dw   = (int)(rest / Nld);
  dst[i] = (k < Kdim && n < Ndim)
             ? (bf16_t)src[((long long)dw * Kdim + k) * Ndim + n]
             : (bf16_t)0.f;
}

__global__ void k_pad_bias(const float* __restrict__ src, float* __restrict__ dst,
                           int Ndim, int Nld) {
  int n = blockIdx.x * blockDim.x + threadIdx.x;
  if (n < Nld) dst[n] = (n < Ndim) ? src[n] : 0.f;
}

// width must be a multiple of 16 (rows 32B aligned)
__global__ void k_rownorm(const bf16_t* __restrict__ X, float* __restrict__ out,
                          int rows, int width) {
  int r = blockIdx.x * blockDim.x + threadIdx.x;
  if (r >= rows) return;
  const v16bf* p = (const v16bf*)(X + (size_t)r * width);
  float s = 0.f;
  for (int c = 0; c < width / 16; ++c) {
    v16bf v = p[c];
#pragma unroll
    for (int j = 0; j < 16; ++j) {
      float f = (float)v[j];
      s += f * f;
    }
  }
  out[r] = s;
}

// ---------------------------------------------------------------------------
// Generic conv-as-GEMM with WMMA bf16 (SAME padding, stride 1, KW in {1,3})
//   Out[m, n] = act( sum_dw sum_k A[b, t+dw-pad, k] * Wt[dw, n, k] + bias[n] )
//   A : [Mtot, lda] bf16 row-major (lda multiple of 32, zero padded K)
//   Wt: [KW, Nld, lda] bf16 transposed weights (zero padded)
// Tile: 128 x 128 per 256-thread block; 8 waves (4 M x 2 N), each wave 32x64
// -> 8 WMMAs per K-step per wave. Software pipelined global->reg->LDS staging.
// ---------------------------------------------------------------------------
#define BM 128
#define BN 128
#define BK 32

__global__ void __launch_bounds__(256)
k_conv_gemm(const bf16_t* __restrict__ A, int lda, int T,
            const bf16_t* __restrict__ Wt, const float* __restrict__ bias,
            bf16_t* __restrict__ Out, int ldo,
            int Mtot, int Nld, int KW, int relu) {
  __shared__ __attribute__((aligned(16))) bf16_t sA [BM][BK + 16]; // 96B row stride
  __shared__ __attribute__((aligned(16))) bf16_t sBT[BN][BK + 16]; // [n][k]

  const int tid  = threadIdx.x;
  const int lane = tid & 31;
  const int wave = tid >> 5;
  const int half = lane >> 4;
  const int l16  = lane & 15;

  const int m0 = blockIdx.y * BM;
  const int n0 = blockIdx.x * BN;
  const int wm = (wave & 3) * 32;   // 4 waves along M (32 rows each)
  const int wn = (wave >> 2) * 64;  // 2 waves along N (64 cols each)
  const int pad = KW >> 1;

  // per-thread A staging coordinates (constant across K loop)
  const int ar  = tid >> 1;          // 0..127
  const int akb = (tid & 1) * 16;    // 0 or 16
  const int am  = m0 + ar;
  const bool arow_ok = am < Mtot;
  int ab = 0, at = 0;
  if (arow_ok) { ab = am / T; at = am - ab * T; }
  const bf16_t* Abase = A + (size_t)ab * T * lda;

  // per-thread B staging coordinates (128 rows x 32 k)
  const int bn  = tid >> 1;          // 0..127
  const int bkb = (tid & 1) * 16;    // 0 or 16
  const bool bok = (n0 + bn) < Nld;  // Nld multiple of 8

  uint4 zz; zz.x = zz.y = zz.z = zz.w = 0u;

  auto stageA = [&](int dw, int k0, uint4& r0, uint4& r1) {
    int ts = at + dw - pad;
    bool ok = arow_ok && (ts >= 0) && (ts < T);
    const uint4* p = (const uint4*)(Abase + (size_t)ts * lda + k0 + akb);
    r0 = ok ? p[0] : zz;
    r1 = ok ? p[1] : zz;
  };
  auto stageB = [&](int dw, int k0, uint4& r0, uint4& r1) {
    const uint4* p = (const uint4*)(Wt + ((size_t)dw * Nld + n0 + bn) * lda + k0 + bkb);
    r0 = bok ? p[0] : zz;
    r1 = bok ? p[1] : zz;
  };

  v8f acc[2][4] = {};
  uint4 ra0, ra1, rb0, rb1;
  stageA(0, 0, ra0, ra1);
  stageB(0, 0, rb0, rb1);

  int dw = 0, k0 = 0;
  for (;;) {
    __syncthreads();
    *(uint4*)&sA[ar][akb]      = ra0;
    *(uint4*)&sA[ar][akb + 8]  = ra1;
    *(uint4*)&sBT[bn][bkb]     = rb0;
    *(uint4*)&sBT[bn][bkb + 8] = rb1;
    __syncthreads();

    // next tile indices
    int ndw = dw, nk0 = k0 + BK;
    if (nk0 >= lda) { nk0 = 0; ndw = dw + 1; }
    const bool have_next = ndw < KW;
    if (have_next) {             // issue next global loads before the WMMAs
      stageA(ndw, nk0, ra0, ra1);
      stageB(ndw, nk0, rb0, rb1);
    }

    // fragments (all b128 LDS loads)
    v16bf af[2], bfg[4];
#pragma unroll
    for (int i = 0; i < 2; ++i) {
      int ra = wm + i * 16 + l16;
      v8bf a0 = *(const v8bf*)&sA[ra][8 * half];
      v8bf a1 = *(const v8bf*)&sA[ra][16 + 8 * half];
      af[i] = __builtin_shufflevector(a0, a1, 0, 1, 2, 3, 4, 5, 6, 7,
                                      8, 9, 10, 11, 12, 13, 14, 15);
    }
#pragma unroll
    for (int j = 0; j < 4; ++j) {
      int cb = wn + j * 16 + l16;
      v8bf b0 = *(const v8bf*)&sBT[cb][8 * half];
      v8bf b1 = *(const v8bf*)&sBT[cb][16 + 8 * half];
      bfg[j] = __builtin_shufflevector(b0, b1, 0, 1, 2, 3, 4, 5, 6, 7,
                                       8, 9, 10, 11, 12, 13, 14, 15);
    }
#pragma unroll
    for (int i = 0; i < 2; ++i)
#pragma unroll
      for (int j = 0; j < 4; ++j)
        acc[i][j] = __builtin_amdgcn_wmma_f32_16x16x32_bf16(
            false, af[i], false, bfg[j], (short)0, acc[i][j], false, false);

    if (!have_next) break;
    dw = ndw; k0 = nk0;
  }

  // epilogue: bias + optional relu, bf16 store
#pragma unroll
  for (int i = 0; i < 2; ++i) {
#pragma unroll
    for (int j = 0; j < 4; ++j) {
      int col = n0 + wn + j * 16 + l16;
      if (col < Nld) {
        float bv = bias[col];
#pragma unroll
        for (int v = 0; v < 8; ++v) {
          int row = m0 + wm + i * 16 + half * 8 + v;
          if (row < Mtot) {
            float val = acc[i][j][v] + bv;
            if (relu) val = fmaxf(val, 0.f);
            Out[(size_t)row * ldo + col] = (bf16_t)val;
          }
        }
      }
    }
  }
}

// ---------------------------------------------------------------------------
// Batched Q·K^T with fused squared-distance epilogue:
//   S[b,t1,t2] = -TEMP * ( q2[b,t1] + k2[b,t2] - 2 * <q, k> )
// Q: [B*T1, Cp], Kf: [B*T2, Cp], Cp = 96 (zero padded)
// ---------------------------------------------------------------------------
__global__ void __launch_bounds__(256)
k_qk_dist(const bf16_t* __restrict__ Q, const bf16_t* __restrict__ Kf,
          const float* __restrict__ q2n, const float* __restrict__ k2n,
          float* __restrict__ S, int T1, int T2, int Cp) {
  __shared__ __attribute__((aligned(16))) bf16_t sA [BM][BK + 16];
  __shared__ __attribute__((aligned(16))) bf16_t sBT[BN][BK + 16];

  const int tid  = threadIdx.x;
  const int lane = tid & 31;
  const int wave = tid >> 5;
  const int half = lane >> 4;
  const int l16  = lane & 15;

  const int b  = blockIdx.z;
  const int m0 = blockIdx.y * BM;
  const int n0 = blockIdx.x * BN;
  const int wm = (wave & 3) * 32;
  const int wn = (wave >> 2) * 64;

  const bf16_t* Qb = Q  + (size_t)b * T1 * Cp;
  const bf16_t* Kb = Kf + (size_t)b * T2 * Cp;

  const int ar  = tid >> 1;
  const int akb = (tid & 1) * 16;
  const bool aok = (m0 + ar) < T1;
  const int bn  = tid >> 1;
  const int bkb = (tid & 1) * 16;
  const bool bok = (n0 + bn) < T2;

  uint4 zz; zz.x = zz.y = zz.z = zz.w = 0u;

  auto stageA = [&](int k0, uint4& r0, uint4& r1) {
    const uint4* p = (const uint4*)(Qb + (size_t)(m0 + ar) * Cp + k0 + akb);
    r0 = aok ? p[0] : zz;
    r1 = aok ? p[1] : zz;
  };
  auto stageB = [&](int k0, uint4& r0, uint4& r1) {
    const uint4* p = (const uint4*)(Kb + (size_t)(n0 + bn) * Cp + k0 + bkb);
    r0 = bok ? p[0] : zz;
    r1 = bok ? p[1] : zz;
  };

  v8f acc[2][4] = {};
  uint4 ra0, ra1, rb0, rb1;
  stageA(0, ra0, ra1);
  stageB(0, rb0, rb1);

  for (int k0 = 0;;) {
    __syncthreads();
    *(uint4*)&sA[ar][akb]      = ra0;
    *(uint4*)&sA[ar][akb + 8]  = ra1;
    *(uint4*)&sBT[bn][bkb]     = rb0;
    *(uint4*)&sBT[bn][bkb + 8] = rb1;
    __syncthreads();

    int nk0 = k0 + BK;
    const bool have_next = nk0 < Cp;
    if (have_next) { stageA(nk0, ra0, ra1); stageB(nk0, rb0, rb1); }

    v16bf af[2], bfg[4];
#pragma unroll
    for (int i = 0; i < 2; ++i) {
      int ra = wm + i * 16 + l16;
      v8bf a0 = *(const v8bf*)&sA[ra][8 * half];
      v8bf a1 = *(const v8bf*)&sA[ra][16 + 8 * half];
      af[i] = __builtin_shufflevector(a0, a1, 0, 1, 2, 3, 4, 5, 6, 7,
                                      8, 9, 10, 11, 12, 13, 14, 15);
    }
#pragma unroll
    for (int j = 0; j < 4; ++j) {
      int cb = wn + j * 16 + l16;
      v8bf b0 = *(const v8bf*)&sBT[cb][8 * half];
      v8bf b1 = *(const v8bf*)&sBT[cb][16 + 8 * half];
      bfg[j] = __builtin_shufflevector(b0, b1, 0, 1, 2, 3, 4, 5, 6, 7,
                                       8, 9, 10, 11, 12, 13, 14, 15);
    }
#pragma unroll
    for (int i = 0; i < 2; ++i)
#pragma unroll
      for (int j = 0; j < 4; ++j)
        acc[i][j] = __builtin_amdgcn_wmma_f32_16x16x32_bf16(
            false, af[i], false, bfg[j], (short)0, acc[i][j], false, false);

    if (!have_next) break;
    k0 = nk0;
  }

#pragma unroll
  for (int i = 0; i < 2; ++i) {
#pragma unroll
    for (int j = 0; j < 4; ++j) {
      int col = n0 + wn + j * 16 + l16;
      if (col < T2) {
        float k2v = k2n[(size_t)b * T2 + col];
#pragma unroll
        for (int v = 0; v < 8; ++v) {
          int row = m0 + wm + i * 16 + half * 8 + v;
          if (row < T1) {
            float d = q2n[(size_t)b * T1 + row] + k2v - 2.f * acc[i][j][v];
            S[((size_t)b * T1 + row) * T2 + col] = -TEMP_C * d;
          }
        }
      }
    }
  }
}

// ---------------------------------------------------------------------------
// Per-row (b,t1): log_softmax over T2=512 + log(prior+1e-8) -> logprob (inplace),
// then masked softmax -> attn. One wave32 per row, 16 cols per lane,
// fully float4/uchar4 vectorized (col = lane*16 + j).
// ---------------------------------------------------------------------------
__global__ void __launch_bounds__(256)
k_softmax(float* __restrict__ logp,          // in: raw scores, out: logprob
          const float* __restrict__ prior,
          const unsigned char* __restrict__ mask,
          float* __restrict__ attn, int B, int T1) {
  constexpr int T2  = 512;
  constexpr int NPL = 16;              // elements per lane
  const int lane = threadIdx.x & 31;
  const int wave = threadIdx.x >> 5;
  const int row  = blockIdx.x * 8 + wave;
  if (row >= B * T1) return;
  const int b = row / T1;
  const size_t base = (size_t)row * T2;

  float4* lp4        = (float4*)(logp + base);
  const float4* pr4  = (const float4*)(prior + base);
  float4* at4        = (float4*)(attn + base);
  const uchar4* mk4  = (const uchar4*)(mask + (size_t)b * T2);

  float s[NPL];
  float mx = -3.0e38f;
#pragma unroll
  for (int q = 0; q < 4; ++q) {
    float4 v = lp4[lane * 4 + q];
    s[4 * q + 0] = v.x; s[4 * q + 1] = v.y; s[4 * q + 2] = v.z; s[4 * q + 3] = v.w;
  }
#pragma unroll
  for (int j = 0; j < NPL; ++j) mx = fmaxf(mx, s[j]);
#pragma unroll
  for (int off = 16; off > 0; off >>= 1) mx = fmaxf(mx, __shfl_xor(mx, off, 32));
  float sum = 0.f;
#pragma unroll
  for (int j = 0; j < NPL; ++j) sum += __expf(s[j] - mx);
#pragma unroll
  for (int off = 16; off > 0; off >>= 1) sum += __shfl_xor(sum, off, 32);
  const float lse = mx + __logf(sum);

  float mx2 = -3.0e38f;
#pragma unroll
  for (int q = 0; q < 4; ++q) {
    float4 pv = pr4[lane * 4 + q];
    uchar4 mk = mk4[lane * 4 + q];
    float lpv[4];
    lpv[0] = s[4 * q + 0] - lse + __logf(pv.x + 1e-8f);
    lpv[1] = s[4 * q + 1] - lse + __logf(pv.y + 1e-8f);
    lpv[2] = s[4 * q + 2] - lse + __logf(pv.z + 1e-8f);
    lpv[3] = s[4 * q + 3] - lse + __logf(pv.w + 1e-8f);
    float4 o; o.x = lpv[0]; o.y = lpv[1]; o.z = lpv[2]; o.w = lpv[3];
    lp4[lane * 4 + q] = o;
    s[4 * q + 0] = mk.x ? lpv[0] : MASK_VAL;
    s[4 * q + 1] = mk.y ? lpv[1] : MASK_VAL;
    s[4 * q + 2] = mk.z ? lpv[2] : MASK_VAL;
    s[4 * q + 3] = mk.w ? lpv[3] : MASK_VAL;
  }
#pragma unroll
  for (int j = 0; j < NPL; ++j) mx2 = fmaxf(mx2, s[j]);
#pragma unroll
  for (int off = 16; off > 0; off >>= 1) mx2 = fmaxf(mx2, __shfl_xor(mx2, off, 32));
  float sum2 = 0.f;
#pragma unroll
  for (int j = 0; j < NPL; ++j) { s[j] = __expf(s[j] - mx2); sum2 += s[j]; }
#pragma unroll
  for (int off = 16; off > 0; off >>= 1) sum2 += __shfl_xor(sum2, off, 32);
  const float inv = 1.f / sum2;
#pragma unroll
  for (int q = 0; q < 4; ++q) {
    float4 o;
    o.x = s[4 * q + 0] * inv; o.y = s[4 * q + 1] * inv;
    o.z = s[4 * q + 2] * inv; o.w = s[4 * q + 3] * inv;
    at4[lane * 4 + q] = o;
  }
}

// ---------------------------------------------------------------------------
// Host-side launcher
// ---------------------------------------------------------------------------
extern "C" void kernel_launch(void* const* d_in, const int* in_sizes, int n_in,
                              void* d_out, int out_size, void* d_ws, size_t ws_size,
                              hipStream_t stream) {
  (void)in_sizes; (void)n_in; (void)out_size; (void)ws_size;

  const int B = 32, T1 = 2048, T2 = 512;
  const int MQ = B * T1;   // 65536
  const int MK = B * T2;   // 16384
  const int CP = 96;       // 80 padded to multiple of 32

  const float* queries = (const float*)d_in[0];
  const int*   keysI   = (const int*)d_in[1];
  const unsigned char* mask = (const unsigned char*)d_in[2];
  const float* prior   = (const float*)d_in[3];
  const float* emb     = (const float*)d_in[4];
  const float* kW1 = (const float*)d_in[5];
  const float* kb1 = (const float*)d_in[6];
  const float* kW2 = (const float*)d_in[7];
  const float* kb2 = (const float*)d_in[8];
  const float* qW1 = (const float*)d_in[9];
  const float* qb1 = (const float*)d_in[10];
  const float* qW2 = (const float*)d_in[11];
  const float* qb2 = (const float*)d_in[12];
  const float* qW3 = (const float*)d_in[13];
  const float* qb3 = (const float*)d_in[14];

  char* wsb = (char*)d_ws;
  size_t off = 0;
  auto bump = [&](size_t bytes) -> void* {
    void* p = (void*)(wsb + off);
    off += (bytes + 255) & ~(size_t)255;
    return p;
  };
  bf16_t* qx   = (bf16_t*)bump((size_t)MQ * CP * 2);
  bf16_t* kx   = (bf16_t*)bump((size_t)MK * 512 * 2);
  bf16_t* wk1b = (bf16_t*)bump((size_t)3 * 1024 * 512 * 2);
  float*  bk1p = (float*)bump(1024 * 4);
  bf16_t* wk2b = (bf16_t*)bump((size_t)96 * 1024 * 2);
  float*  bk2p = (float*)bump(96 * 4);
  bf16_t* wq1b = (bf16_t*)bump((size_t)3 * 160 * 96 * 2);
  float*  bq1p = (float*)bump(160 * 4);
  bf16_t* wq2b = (bf16_t*)bump((size_t)96 * 160 * 2);
  float*  bq2p = (float*)bump(96 * 4);
  bf16_t* wq3b = (bf16_t*)bump((size_t)96 * 96 * 2);
  float*  bq3p = (float*)bump(96 * 4);
  bf16_t* hk1  = (bf16_t*)bump((size_t)MK * 1024 * 2);
  bf16_t* kf   = (bf16_t*)bump((size_t)MK * CP * 2);
  bf16_t* hq1  = (bf16_t*)bump((size_t)MQ * 160 * 2);
  bf16_t* hq2  = (bf16_t*)bump((size_t)MQ * CP * 2);
  bf16_t* qf   = (bf16_t*)bump((size_t)MQ * CP * 2);
  float*  q2n  = (float*)bump((size_t)MQ * 4);
  float*  k2n  = (float*)bump((size_t)MK * 4);

  // --- prep ---
  {
    long long n = (long long)MQ * CP;
    k_cast_pad_rows<<<(unsigned)((n + 255) / 256), 256, 0, stream>>>(queries, qx, MQ, 80, CP);
  }
  {
    long long n = (long long)MK * 512;
    k_embed<<<(unsigned)((n + 255) / 256), 256, 0, stream>>>(keysI, emb, kx, MK, 512);
  }
  auto castW = [&](const float* s, bf16_t* d, int KW, int Kd, int Kp, int Nd, int Nl) {
    long long n = (long long)KW * Nl * Kp;
    k_cast_weight_t<<<(unsigned)((n + 255) / 256), 256, 0, stream>>>(s, d, KW, Kd, Kp, Nd, Nl);
  };
  auto padB = [&](const float* s, float* d, int Nd, int Nl) {
    k_pad_bias<<<(Nl + 255) / 256, 256, 0, stream>>>(s, d, Nd, Nl);
  };
  castW(kW1, wk1b, 3, 512, 512, 1024, 1024);  padB(kb1, bk1p, 1024, 1024);
  castW(kW2, wk2b, 1, 1024, 1024, 80, 96);    padB(kb2, bk2p, 80, 96);
  castW(qW1, wq1b, 3, 80, 96, 160, 160);      padB(qb1, bq1p, 160, 160);
  castW(qW2, wq2b, 1, 160, 160, 80, 96);      padB(qb2, bq2p, 80, 96);
  castW(qW3, wq3b, 1, 80, 96, 80, 96);        padB(qb3, bq3p, 80, 96);

  // --- key path ---
  k_conv_gemm<<<dim3(1024 / BN, MK / BM), 256, 0, stream>>>(
      kx, 512, T2, wk1b, bk1p, hk1, 1024, MK, 1024, 3, 1);
  k_conv_gemm<<<dim3(1, MK / BM), 256, 0, stream>>>(
      hk1, 1024, T2, wk2b, bk2p, kf, 96, MK, 96, 1, 0);

  // --- query path ---
  k_conv_gemm<<<dim3(2, MQ / BM), 256, 0, stream>>>(
      qx, 96, T1, wq1b, bq1p, hq1, 160, MQ, 160, 3, 1);
  k_conv_gemm<<<dim3(1, MQ / BM), 256, 0, stream>>>(
      hq1, 160, T1, wq2b, bq2p, hq2, 96, MQ, 96, 1, 1);
  k_conv_gemm<<<dim3(1, MQ / BM), 256, 0, stream>>>(
      hq2, 96, T1, wq3b, bq3p, qf, 96, MQ, 96, 1, 0);

  // --- norms ---
  k_rownorm<<<(MQ + 255) / 256, 256, 0, stream>>>(qf, q2n, MQ, CP);
  k_rownorm<<<(MK + 255) / 256, 256, 0, stream>>>(kf, k2n, MK, CP);

  // --- QK distance into logprob half of d_out, then fused softmaxes ---
  float* out_attn = (float*)d_out;
  float* out_logp = out_attn + (size_t)B * T1 * T2;
  k_qk_dist<<<dim3(T2 / BN, T1 / BM, B), 256, 0, stream>>>(
      qf, kf, q2n, k2n, out_logp, T1, T2, CP);
  k_softmax<<<(B * T1) / 8, 256, 0, stream>>>(out_logp, prior, mask, out_attn, B, T1);
}
